// AttenLayer_27281632264963
// MI455X (gfx1250) — compile-verified, matrix operand order checked by reading, not run on hardware
//
#include <hip/hip_runtime.h>
#include <math.h>

// ---------------- problem constants (from reference) ----------------
#define HIDDEN 512   // K of proj GEMM
#define ATTEN  512   // N of proj GEMM
#define BATCH  32
#define STEP   2048
#define MROWS  (BATCH * STEP)          // 65536 rows of the proj GEMM
#define MTILE  64                      // rows per block in the scores kernel
#define SCH    32                      // s-chunks in the weighted-sum pass

// ---------------- WMMA vector types (gfx1250, wave32) ---------------
typedef __attribute__((ext_vector_type(16))) __bf16 v16bf;
typedef __attribute__((ext_vector_type(8)))  __bf16 v8bf;
typedef __attribute__((ext_vector_type(8)))  float  v8f;

// A-tile staged in LDS: MTILE rows x 512 K of bf16, row stride 520 bf16
// (1040 B = 65*16 B -> every ds_load_b128 fragment address is 16B aligned,
//  and 260 dwords/row -> 4-bank shift per row -> conflict-free b128 reads).
#define K_STRIDE 520

// ====================================================================
// Kernel 0: convert W [H=512 (k), A=512 (n)] f32 -> bf16, pre-swizzled
// into WMMA B-fragment order so each lane's 16 bf16 are contiguous:
//   flat = ((nt*16 + kt)*32 + lane)*16 + e
//   n    = nt*16 + (lane & 15), khalf = lane >> 4
//   K    = kt*32 + (e<8 ? khalf*8+e : 16 + khalf*8 + (e-8))
// (16-bit B-matrix 32x16 layout per CDNA5 ISA 7.12.2)
// ====================================================================
__global__ void atten_prep_w(const float* __restrict__ W, __bf16* __restrict__ Wb) {
    int o = blockIdx.x * 256 + threadIdx.x;   // 0 .. 512*512-1
    int e  = o & 15;
    int ln = (o >> 4) & 31;
    int kt = (o >> 9) & 15;
    int nt = o >> 13;
    int n  = nt * 16 + (ln & 15);
    int kh = ln >> 4;
    int K  = kt * 32 + ((e < 8) ? (kh * 8 + e) : (16 + kh * 8 + (e - 8)));
    Wb[o] = (__bf16)W[K * ATTEN + n];
}

// ====================================================================
// Kernel 1: fused  scores[row] = sum_n tanh( x_row . W[:,n] + b[n] ) * ctx[n]
// Block = 256 threads (8 waves), 64 rows per block (1024 blocks re-stream
// the 512KB swizzled W from L2 -> 512MB total L2 traffic).
//   wave -> (rowSub = wv>>1 in 0..3 : which 16-row WMMA tile,
//            nHalf  = wv&1         : n-tiles [nHalf*16, nHalf*16+16))
// Two passes of 8 independent bf16 WMMA accumulators per wave.
// B addressing: one 32-bit element offset per kt (< 2^21, exact in int),
// fragment i at a compile-time i*16KB step -> saddr + voffset + immediate.
// ====================================================================
__global__ void __launch_bounds__(256)
atten_scores(const float* __restrict__ X,      // [MROWS, HIDDEN] f32
             const __bf16* __restrict__ Wb,    // swizzled bf16 W
             const float* __restrict__ bias,   // [ATTEN]
             const float* __restrict__ ctx,    // [ATTEN]
             float* __restrict__ scores)       // [MROWS]
{
    __shared__ __bf16 As[MTILE * K_STRIDE];    // ~65 KB of the 320 KB WGP LDS
    __shared__ float  scoreBuf[MTILE];

    const int t  = threadIdx.x;
    const int m0 = blockIdx.x * MTILE;

    if (t < MTILE) scoreBuf[t] = 0.0f;

    // ---- stage 64 rows x 512 f32 -> bf16 LDS tile (each thread: 32 float4)
    {
        const int r  = t >> 2;                 // 0..63 row
        const int c0 = t & 3;                  // float4 slot within row group
        const float4* Xr = (const float4*)(X + (size_t)(m0 + r) * HIDDEN);
        #pragma unroll
        for (int i = 0; i < 32; ++i) {
            int c4 = c0 + i * 4;               // 0..127 float4 index
            float4 v = Xr[c4];
            int o = r * K_STRIDE + c4 * 4;
            As[o + 0] = (__bf16)v.x;
            As[o + 1] = (__bf16)v.y;
            As[o + 2] = (__bf16)v.z;
            As[o + 3] = (__bf16)v.w;
        }
    }
    __syncthreads();

    const int wv     = t >> 5;
    const int lane   = t & 31;
    const int rowSub = wv >> 1;      // 0..3 -> rows [rowSub*16, +16)
    const int nHalf  = wv & 1;       // 0..1 -> n-tiles [nHalf*16, +16)
    const int m      = lane & 15;
    const int half   = lane >> 4;

    const __bf16* Arow    = &As[(rowSub * 16 + m) * K_STRIDE + half * 8];
    const int     laneOff = lane << 4;         // lane*16 elements (32B/lane)

    float sAcc[8];
    #pragma unroll
    for (int j = 0; j < 8; ++j) sAcc[j] = 0.0f;

    for (int p = 0; p < 2; ++p) {              // two groups of 8 n-tiles
        const int ntBase = nHalf * 16 + p * 8;
        v8f acc[8] = {};

        for (int kt = 0; kt < 16; ++kt) {
            // A fragment: two 16B runs (K = kt*32 + half*8 and +16 further)
            const v8bf* pa = (const v8bf*)(Arow + kt * 32);
            v8bf lo = pa[0];
            v8bf hi = pa[2];
            v16bf a;
            #pragma unroll
            for (int j = 0; j < 8; ++j) { a[j] = lo[j]; a[j + 8] = hi[j]; }

            // one 32-bit element offset per kt; fragments step by i*8192 elem
            const int bOff = ((ntBase * 16 + kt) << 9) + laneOff;
            const __bf16* pkt = Wb + bOff;

            #pragma unroll
            for (int i = 0; i < 8; ++i) {
                v16bf bfr = *(const v16bf*)(pkt + (i << 13));  // imm = i*16KB
                acc[i] = __builtin_amdgcn_wmma_f32_16x16x32_bf16(
                             false, a, false, bfr, (short)0, acc[i], false, false);
            }
        }

        // ---- epilogue: tanh(+bias) * context
        #pragma unroll
        for (int i = 0; i < 8; ++i) {
            int n  = (ntBase + i) * 16 + (lane & 15);   // this lane's column
            float bb = bias[n];
            float cc = ctx[n];
            #pragma unroll
            for (int j = 0; j < 8; ++j)
                sAcc[j] += tanhf(acc[i][j] + bb) * cc;
        }
    }

    // butterfly over the 16 lanes that share this row set (stays in half)
    #pragma unroll
    for (int off = 1; off < 16; off <<= 1) {
        #pragma unroll
        for (int j = 0; j < 8; ++j)
            sAcc[j] += __shfl_xor(sAcc[j], off, 32);
    }

    if ((lane & 15) == 0) {
        int rb = rowSub * 16 + half * 8;     // local rows rb..rb+7
        #pragma unroll
        for (int j = 0; j < 8; ++j)
            atomicAdd(&scoreBuf[rb + j], sAcc[j]);   // combine the 2 n-halves
    }
    __syncthreads();

    if (t < MTILE) scores[m0 + t] = scoreBuf[t];
}

// ====================================================================
// Kernel 2: masked softmax per batch, in place (padded lanes -> exactly 0)
// ====================================================================
__global__ void __launch_bounds__(256)
atten_softmax(const long long* __restrict__ lens, float* __restrict__ sc) {
    const int b = blockIdx.x, t = threadIdx.x;
    const int len = (int)lens[b];
    float* row = sc + b * STEP;

    __shared__ float red[256];

    float v[8];
    float mx = -INFINITY;
    #pragma unroll
    for (int i = 0; i < 8; ++i) {
        int s = t + i * 256;
        float x = (s < len) ? row[s] : -INFINITY;
        v[i] = x;
        mx = fmaxf(mx, x);
    }
    red[t] = mx; __syncthreads();
    for (int o = 128; o > 0; o >>= 1) {
        if (t < o) red[t] = fmaxf(red[t], red[t + o]);
        __syncthreads();
    }
    const float M = red[0]; __syncthreads();

    float sum = 0.0f;
    #pragma unroll
    for (int i = 0; i < 8; ++i) {
        float e = (v[i] == -INFINITY) ? 0.0f : __expf(v[i] - M);
        v[i] = e;
        sum += e;
    }
    red[t] = sum; __syncthreads();
    for (int o = 128; o > 0; o >>= 1) {
        if (t < o) red[t] += red[t + o];
        __syncthreads();
    }
    const float inv = 1.0f / red[0];

    #pragma unroll
    for (int i = 0; i < 8; ++i)
        row[t + i * 256] = v[i] * inv;
}

// ====================================================================
// Kernel 3a: partial weighted sums, S split into SCH chunks so enough
// float4 loads are in flight to reach the HBM roof (1024 blocks x 128
// threads x 16B). Chunks fully past `len` write zeros (weights there are
// already exactly 0, so the in-boundary chunk needs no per-s mask).
// ====================================================================
__global__ void __launch_bounds__(128)
atten_wsum_part(const float* __restrict__ X, const float* __restrict__ w,
                const long long* __restrict__ lens, float* __restrict__ part) {
    const int b  = blockIdx.y;
    const int ch = blockIdx.x;
    const int s0 = ch * (STEP / SCH);                  // 64 s per chunk
    const int len = (int)lens[b];

    int sEnd = len - s0;
    if (sEnd > (STEP / SCH)) sEnd = STEP / SCH;

    const int h = threadIdx.x * 4;
    const float* xb = X + ((size_t)b * STEP + s0) * HIDDEN + h;
    const float* wb = w + b * STEP + s0;

    float4 acc = make_float4(0.f, 0.f, 0.f, 0.f);
    #pragma unroll 4
    for (int s = 0; s < sEnd; ++s) {
        float  ww = wb[s];                              // wave-uniform scalar
        float4 x4 = *(const float4*)(xb + (size_t)s * HIDDEN);
        acc.x += ww * x4.x;
        acc.y += ww * x4.y;
        acc.z += ww * x4.z;
        acc.w += ww * x4.w;
    }
    *(float4*)(part + ((size_t)(b * SCH + ch)) * HIDDEN + h) = acc;
}

// ====================================================================
// Kernel 3b: fixed-order chunk reduction -> out[b,h] (deterministic)
// ====================================================================
__global__ void __launch_bounds__(256)
atten_wsum_reduce(const float* __restrict__ part, float* __restrict__ out) {
    const int b = blockIdx.x, t = threadIdx.x;
    for (int h = t; h < HIDDEN; h += 256) {
        float acc = 0.0f;
        #pragma unroll
        for (int c = 0; c < SCH; ++c)
            acc += part[((size_t)(b * SCH + c)) * HIDDEN + h];
        out[b * HIDDEN + h] = acc;
    }
}

// ====================================================================
// kernel_launch
//   inputs: 0 nn_outs f32[32*2048*512], 1 batch_lens i64[32],
//           2 W f32[512*512], 3 b f32[512], 4 context f32[512]
//   out:    f32[32*512]
//   ws:  [0, 512KB)      W_bf16 swizzled
//        [512KB, 768KB)  scores/atten f32 [B*S]
//        [768KB, 2.75MB) partial weighted sums f32 [B, SCH, H]
// ====================================================================
extern "C" void kernel_launch(void* const* d_in, const int* in_sizes, int n_in,
                              void* d_out, int out_size, void* d_ws, size_t ws_size,
                              hipStream_t stream) {
    (void)in_sizes; (void)n_in; (void)out_size; (void)ws_size;

    const float*     nn_outs = (const float*)d_in[0];
    const long long* lens    = (const long long*)d_in[1];   // int64 per reference
    const float*     W       = (const float*)d_in[2];
    const float*     bias    = (const float*)d_in[3];
    const float*     ctx     = (const float*)d_in[4];
    float*           out     = (float*)d_out;

    __bf16* Wb     = (__bf16*)d_ws;                                 // 512 KB
    float*  scores = (float*)((char*)d_ws + (size_t)512 * 1024);    // 256 KB
    float*  part   = (float*)((char*)d_ws + (size_t)768 * 1024);    // 2 MB

    atten_prep_w     <<<(HIDDEN * ATTEN) / 256, 256, 0, stream>>>(W, Wb);
    atten_scores     <<<MROWS / MTILE, 256, 0, stream>>>(nn_outs, Wb, bias, ctx, scores);
    atten_softmax    <<<BATCH, 256, 0, stream>>>(lens, scores);
    atten_wsum_part  <<<dim3(SCH, BATCH), 128, 0, stream>>>(nn_outs, scores, lens, part);
    atten_wsum_reduce<<<BATCH, 256, 0, stream>>>(part, out);
}
